// KGEncoder_50087908606359
// MI455X (gfx1250) — matrix operand order repaired and checked by reading, not compiled, not run on hardware
//
#include <hip/hip_runtime.h>
#include <math.h>

// ---------------------------------------------------------------------------
// HGT encoder for MI455X (gfx1250, wave32).
// Dense projections (128x128) and per-head 32x32 relation transforms use
// v_wmma_f32_16x16x32_bf16. Weights are pre-packed once per GEMM into the
// WMMA B-fragment layout (bf16), so each wave fetches its fragment as a
// lane-contiguous coalesced b128 load. A-fragments use float4 (b128) loads.
// Edge attention (gather / segment-softmax / scatter) is VMEM+atomic bound.
// ---------------------------------------------------------------------------

typedef __attribute__((ext_vector_type(16))) __bf16 v16bf;
typedef __attribute__((ext_vector_type(8)))  float  v8f;

#define C_DIM 128
#define H_DIM 4
#define D_DIM 32

// ---------------- weight packing into WMMA B-fragment layout ---------------
// Wp frag index f = nt*4 + kk (nt: 16-col tile 0..7, kk: K-chunk 0..3).
// Lane l's 16 bf16 values stored contiguously: element idx -> K = kk*32 +
// (l>>4)*8 + (idx&7) + 16*(idx>>3), col = nt*16 + (l&15).
__global__ void pack_w_kernel(const float* __restrict__ W, __bf16* __restrict__ Wp)
{
    const int t = blockIdx.x * blockDim.x + threadIdx.x;   // 0..16383
    if (t >= 8 * 4 * 32 * 16) return;
    const int idx = t & 15, l = (t >> 4) & 31, kk = (t >> 9) & 3, nt = (t >> 11) & 7;
    const int k   = kk * 32 + (l >> 4) * 8 + 16 * (idx >> 3) + (idx & 7);
    const int col = nt * 16 + (l & 15);
    Wp[t] = (__bf16)W[k * C_DIM + col];
}

// Rp frag index f = h*2 + nh (head h 0..3, 16-col half nh 0..1); K=32.
__global__ void pack_r_kernel(const float* __restrict__ R, __bf16* __restrict__ Rp)
{
    const int t = blockIdx.x * blockDim.x + threadIdx.x;   // 0..4095
    if (t >= 4 * 2 * 32 * 16) return;
    const int idx = t & 15, l = (t >> 4) & 31, nh = (t >> 9) & 1, h = (t >> 10) & 3;
    const int k   = (l >> 4) * 8 + 16 * (idx >> 3) + (idx & 7);
    const int col = nh * 16 + (l & 15);
    Rp[t] = (__bf16)R[h * (D_DIM * D_DIM) + k * D_DIM + col];
}

// ---------------------------- WMMA GEMM ------------------------------------
// Y[M,128] = X[M,128] @ W[128,128] + bias[128]; W pre-packed (Wp).
// One wave per 16-row tile; 8 column tiles; K fully unrolled (4 steps).
__global__ __launch_bounds__(32)
void gemm_bias_wmma(const float* __restrict__ X, const __bf16* __restrict__ Wp,
                    const float* __restrict__ bias, float* __restrict__ Y, int M)
{
    const int lane  = threadIdx.x;
    const int tileM = blockIdx.x;
    const int row   = tileM * 16 + (lane & 15);
    const int kh    = (lane >> 4) * 8;        // A-frag K-half offset
    const bool rok  = row < M;
    const float4* xr4 = (const float4*)(X + (size_t)row * C_DIM);
    const v16bf*  Wv  = (const v16bf*)Wp;

    v8f acc[8];
    #pragma unroll
    for (int nt = 0; nt < 8; ++nt)
        acc[nt] = (v8f){0.f,0.f,0.f,0.f,0.f,0.f,0.f,0.f};

    #pragma unroll
    for (int kk = 0; kk < 4; ++kk) {
        v16bf a;
        if (rok) {
            const int b0 = (kk * 32 + kh) >> 2;
            const float4 p0 = xr4[b0],     p1 = xr4[b0 + 1];
            const float4 p2 = xr4[b0 + 4], p3 = xr4[b0 + 5];
            a[0]=(__bf16)p0.x;  a[1]=(__bf16)p0.y;  a[2]=(__bf16)p0.z;  a[3]=(__bf16)p0.w;
            a[4]=(__bf16)p1.x;  a[5]=(__bf16)p1.y;  a[6]=(__bf16)p1.z;  a[7]=(__bf16)p1.w;
            a[8]=(__bf16)p2.x;  a[9]=(__bf16)p2.y;  a[10]=(__bf16)p2.z; a[11]=(__bf16)p2.w;
            a[12]=(__bf16)p3.x; a[13]=(__bf16)p3.y; a[14]=(__bf16)p3.z; a[15]=(__bf16)p3.w;
        } else {
            #pragma unroll
            for (int i = 0; i < 16; ++i) a[i] = (__bf16)0.f;
        }
        #pragma unroll
        for (int nt = 0; nt < 8; ++nt) {
            const v16bf b = Wv[(nt * 4 + kk) * 32 + lane];   // coalesced b128 pair
            acc[nt] = __builtin_amdgcn_wmma_f32_16x16x32_bf16(
                false, a, false, b, (short)0, acc[nt], false, false);
        }
    }
    const int mbase = tileM * 16 + 8 * (lane >> 4);
    #pragma unroll
    for (int nt = 0; nt < 8; ++nt) {
        const int col = nt * 16 + (lane & 15);
        const float bv = bias ? bias[col] : 0.f;
        #pragma unroll
        for (int r = 0; r < 8; ++r) {
            const int gm = mbase + r;
            if (gm < M) Y[(size_t)gm * C_DIM + col] = acc[nt][r] + bv;
        }
    }
}

// ---------------- per-head relation transform (batched 32x32 GEMM) ---------
// Y[n, h*32+e] = sum_d X[n, h*32+d] * R[h, d, e]; R pre-packed (Rp).
__global__ __launch_bounds__(32)
void rel_transform_wmma(const float* __restrict__ X, const __bf16* __restrict__ Rp,
                        float* __restrict__ Y, int M)
{
    const int lane  = threadIdx.x;
    const int tileM = blockIdx.x;
    const int h     = blockIdx.y >> 1;
    const int nh    = blockIdx.y & 1;
    const int row   = tileM * 16 + (lane & 15);
    const int kh    = (lane >> 4) * 8;
    const bool rok  = row < M;
    const float4* xr4 = (const float4*)(X + (size_t)row * C_DIM + h * D_DIM);

    v16bf a;
    if (rok) {
        const int b0 = kh >> 2;
        const float4 p0 = xr4[b0],     p1 = xr4[b0 + 1];
        const float4 p2 = xr4[b0 + 4], p3 = xr4[b0 + 5];
        a[0]=(__bf16)p0.x;  a[1]=(__bf16)p0.y;  a[2]=(__bf16)p0.z;  a[3]=(__bf16)p0.w;
        a[4]=(__bf16)p1.x;  a[5]=(__bf16)p1.y;  a[6]=(__bf16)p1.z;  a[7]=(__bf16)p1.w;
        a[8]=(__bf16)p2.x;  a[9]=(__bf16)p2.y;  a[10]=(__bf16)p2.z; a[11]=(__bf16)p2.w;
        a[12]=(__bf16)p3.x; a[13]=(__bf16)p3.y; a[14]=(__bf16)p3.z; a[15]=(__bf16)p3.w;
    } else {
        #pragma unroll
        for (int i = 0; i < 16; ++i) a[i] = (__bf16)0.f;
    }
    const v16bf b = ((const v16bf*)Rp)[(h * 2 + nh) * 32 + lane];
    v8f acc = (v8f){0.f,0.f,0.f,0.f,0.f,0.f,0.f,0.f};
    acc = __builtin_amdgcn_wmma_f32_16x16x32_bf16(
        false, a, false, b, (short)0, acc, false, false);

    const int mbase = tileM * 16 + 8 * (lane >> 4);
    const int gcol  = h * D_DIM + nh * 16 + (lane & 15);
    #pragma unroll
    for (int r = 0; r < 8; ++r) {
        const int gm = mbase + r;
        if (gm < M) Y[(size_t)gm * C_DIM + gcol] = acc[r];
    }
}

// -------------------------- elementwise / edge kernels ---------------------
__global__ void lin_in_relu_kernel(const float* __restrict__ X, const float* __restrict__ W,
                                   const float* __restrict__ B, float* __restrict__ Y, int M)
{
    const long long t = (long long)blockIdx.x * blockDim.x + threadIdx.x;
    if (t >= (long long)M * C_DIM) return;
    const int n = (int)(t >> 7), c = (int)(t & 127);
    float s = B[c];
    const float* x = X + (size_t)n * 16;
    #pragma unroll
    for (int i = 0; i < 16; ++i) s += x[i] * W[i * C_DIM + c];
    Y[t] = fmaxf(s, 0.f);
}

__global__ void fill_f32(float* __restrict__ p, float v, long long n)
{
    const long long t = (long long)blockIdx.x * blockDim.x + threadIdx.x;
    if (t < n) p[t] = v;
}

__device__ inline void atomicMaxF(float* addr, float val)
{
    if (val >= 0.f) atomicMax((int*)addr, __float_as_int(val));
    else            atomicMin((unsigned int*)addr, __float_as_uint(val));
}

__global__ void edge_alpha_kernel(const float* __restrict__ q, const float* __restrict__ krel,
                                  const int* __restrict__ sidx, const int* __restrict__ didx,
                                  const float* __restrict__ prel,
                                  float* __restrict__ alpha, float* __restrict__ amax, int E)
{
    const long long t = (long long)blockIdx.x * blockDim.x + threadIdx.x;
    if (t >= (long long)E * H_DIM) return;
    const int e = (int)(t >> 2), h = (int)(t & 3);
    const int s = sidx[e], d = didx[e];
    const float4* qp = (const float4*)(q    + (size_t)d * C_DIM + h * D_DIM);
    const float4* kp = (const float4*)(krel + (size_t)s * C_DIM + h * D_DIM);
    float dot = 0.f;
    #pragma unroll
    for (int j = 0; j < 8; ++j) {
        const float4 a = qp[j], b = kp[j];
        dot += a.x * b.x + a.y * b.y + a.z * b.z + a.w * b.w;
    }
    const float a = dot * prel[h] * 0.17677669529663687f;   // 1/sqrt(32)
    alpha[t] = a;
    atomicMaxF(&amax[d * H_DIM + h], a);
}

__global__ void edge_exp_kernel(const int* __restrict__ didx, const float* __restrict__ amax,
                                float* __restrict__ alpha, float* __restrict__ asum, int E)
{
    const long long t = (long long)blockIdx.x * blockDim.x + threadIdx.x;
    if (t >= (long long)E * H_DIM) return;
    const int e = (int)(t >> 2), h = (int)(t & 3);
    const int d = didx[e];
    float m = amax[d * H_DIM + h];
    if (!__builtin_isfinite(m)) m = 0.f;
    const float ex = expf(alpha[t] - m);
    alpha[t] = ex;
    atomicAdd(&asum[d * H_DIM + h], ex);
}

// one thread per (edge, 4-channel group): float4 value load + 4 atomics
__global__ void edge_msg_kernel(const int* __restrict__ sidx, const int* __restrict__ didx,
                                const float* __restrict__ vrel, const float* __restrict__ alpha,
                                const float* __restrict__ asum, float* __restrict__ out,
                                long long total /* E*32 */)
{
    const long long t = (long long)blockIdx.x * blockDim.x + threadIdx.x;
    if (t >= total) return;
    const int e = (int)(t >> 5), c4 = (int)(t & 31), h = c4 >> 3;
    const int s = sidx[e], d = didx[e];
    const float attn = alpha[(size_t)e * H_DIM + h] / (asum[d * H_DIM + h] + 1e-16f);
    const float4 v = ((const float4*)vrel)[(size_t)s * 32 + c4];
    float* op = out + (size_t)d * C_DIM + c4 * 4;
    atomicAdd(op + 0, v.x * attn);
    atomicAdd(op + 1, v.y * attn);
    atomicAdd(op + 2, v.z * attn);
    atomicAdd(op + 3, v.w * attn);
}

__global__ void gelu_ip_kernel(float* __restrict__ x, long long n)
{
    const long long t = (long long)blockIdx.x * blockDim.x + threadIdx.x;
    if (t >= n) return;
    const float v = x[t];
    x[t] = 0.5f * v * (1.f + erff(v * 0.70710678118654752f));
}

__global__ void skip_combine_kernel(float* __restrict__ x, const float* __restrict__ o,
                                    const float* __restrict__ skip, long long n)
{
    const long long t = (long long)blockIdx.x * blockDim.x + threadIdx.x;
    if (t >= n) return;
    const float a = 1.f / (1.f + expf(-skip[0]));
    x[t] = a * o[t] + (1.f - a) * x[t];
}

__global__ void gather2_kernel(const float* __restrict__ rep, const int* __restrict__ na,
                               const int* __restrict__ nb, float* __restrict__ out, int N)
{
    const long long t = (long long)blockIdx.x * blockDim.x + threadIdx.x;
    if (t >= (long long)N * 32) return;
    const int i = (int)(t >> 5), c = (int)(t & 31);
    const float4* rp = (const float4*)rep;
    float4* op = (float4*)out;
    op[t]                    = rp[(size_t)na[i] * 32 + c];
    op[(size_t)N * 32 + t]   = rp[(size_t)nb[i] * 32 + c];
}

// ------------------------------- host side ---------------------------------
extern "C" void kernel_launch(void* const* d_in, const int* in_sizes, int n_in,
                              void* d_out, int out_size, void* d_ws, size_t ws_size,
                              hipStream_t stream)
{
    const float* x_g_in = (const float*)d_in[0];
    const float* x_d_in = (const float*)d_in[1];
    const int* gg_src = (const int*)d_in[2]; const int* gg_dst = (const int*)d_in[3];
    const int* dg_src = (const int*)d_in[4]; const int* dg_dst = (const int*)d_in[5];
    const int* gd_src = (const int*)d_in[6]; const int* gd_dst = (const int*)d_in[7];
    const int* nodea  = (const int*)d_in[8]; const int* nodeb  = (const int*)d_in[9];

    const int NG  = in_sizes[0] / 16;
    const int ND  = in_sizes[1] / 16;
    const int Egg = in_sizes[2];
    const int Edg = in_sizes[4];
    const int Egd = in_sizes[6];
    const int NP  = in_sizes[8];

    // params pytree leaves, alphabetical dict-key (jax) flatten order
    const int PB = 10;
    enum { A_D_B=0, A_D_W, A_G_B, A_G_W, AREL_DG, AREL_GD, AREL_GG,
           K_D_B, K_D_W, K_G_B, K_G_W, MREL_DG, MREL_GD, MREL_GG,
           PREL_DG, PREL_GD, PREL_GG, Q_D_B, Q_D_W, Q_G_B, Q_G_W,
           SKIP_D, SKIP_G, V_D_B, V_D_W, V_G_B, V_G_W };
    auto P = [&](int layer, int k) -> const float* {
        return (const float*)d_in[PB + 27 * layer + k];
    };
    const float* LIN_IN_D_B = (const float*)d_in[PB + 54];
    const float* LIN_IN_D_W = (const float*)d_in[PB + 55];
    const float* LIN_IN_G_B = (const float*)d_in[PB + 56];
    const float* LIN_IN_G_W = (const float*)d_in[PB + 57];
    const float* LIN_OUT_B  = (const float*)d_in[PB + 58];
    const float* LIN_OUT_W  = (const float*)d_in[PB + 59];

    // workspace carve: bf16 packed-weight buffers first (alignment), then f32
    __bf16* Wp = (__bf16*)d_ws;                 // 16384 bf16 = 32 KB
    __bf16* Rp = Wp + 16384;                    //  4096 bf16 =  8 KB
    float* ws  = (float*)(Rp + 4096);           // 16B-aligned (40960 bytes in)
    const size_t NGf = (size_t)NG * C_DIM, NDf = (size_t)ND * C_DIM;
    size_t off = 0;
    float* xg   = ws + off; off += NGf;
    float* xd   = ws + off; off += NDf;
    float* kg   = ws + off; off += NGf;
    float* qg   = ws + off; off += NGf;
    float* vg   = ws + off; off += NGf;
    float* kd   = ws + off; off += NDf;
    float* qd   = ws + off; off += NDf;
    float* vd   = ws + off; off += NDf;
    float* krel = ws + off; off += NGf;   // sized for max-src (g)
    float* vrel = ws + off; off += NGf;
    float* outg = ws + off; off += NGf;
    float* outd = ws + off; off += NDf;
    float* alpha= ws + off; off += (size_t)Egg * H_DIM;
    float* amax = ws + off; off += (size_t)NG * H_DIM;
    float* asum = ws + off; off += (size_t)NG * H_DIM;

    auto nblk = [](long long n, int t) { return (unsigned)((n + t - 1) / t); };

    auto gemm = [&](const float* X, const float* W, const float* B, float* Y, int M) {
        pack_w_kernel<<<64, 256, 0, stream>>>(W, Wp);
        gemm_bias_wmma<<<dim3(nblk(M, 16)), dim3(32), 0, stream>>>(X, Wp, B, Y, M);
    };
    auto relT = [&](const float* X, const float* R, float* Y, int M) {
        pack_r_kernel<<<16, 256, 0, stream>>>(R, Rp);
        rel_transform_wmma<<<dim3(nblk(M, 16), 8), dim3(32), 0, stream>>>(X, Rp, Y, M);
    };
    auto relation = [&](const float* ksrc, const float* vsrc, int Nsrc,
                        const float* qdst, float* outdst, int Ndst,
                        const int* si, const int* di, int E,
                        const float* arel, const float* mrel, const float* prel) {
        relT(ksrc, arel, krel, Nsrc);
        relT(vsrc, mrel, vrel, Nsrc);
        const long long nm = (long long)Ndst * H_DIM;
        fill_f32<<<nblk(nm, 256), 256, 0, stream>>>(amax, -__builtin_inff(), nm);
        fill_f32<<<nblk(nm, 256), 256, 0, stream>>>(asum, 0.f, nm);
        const long long eh = (long long)E * H_DIM;
        edge_alpha_kernel<<<nblk(eh, 256), 256, 0, stream>>>(qdst, krel, si, di, prel, alpha, amax, E);
        edge_exp_kernel<<<nblk(eh, 256), 256, 0, stream>>>(di, amax, alpha, asum, E);
        const long long ec = (long long)E * 32;
        edge_msg_kernel<<<nblk(ec, 256), 256, 0, stream>>>(si, di, vrel, alpha, asum, outdst, ec);
    };

    // ---- input projection + relu ----
    lin_in_relu_kernel<<<nblk((long long)NG * C_DIM, 256), 256, 0, stream>>>(
        x_g_in, LIN_IN_G_W, LIN_IN_G_B, xg, NG);
    lin_in_relu_kernel<<<nblk((long long)ND * C_DIM, 256), 256, 0, stream>>>(
        x_d_in, LIN_IN_D_W, LIN_IN_D_B, xd, ND);

    // ---- HGT layers ----
    for (int l = 0; l < 2; ++l) {
        gemm(xg, P(l, K_G_W), P(l, K_G_B), kg, NG);
        gemm(xg, P(l, Q_G_W), P(l, Q_G_B), qg, NG);
        gemm(xg, P(l, V_G_W), P(l, V_G_B), vg, NG);
        gemm(xd, P(l, K_D_W), P(l, K_D_B), kd, ND);
        gemm(xd, P(l, Q_D_W), P(l, Q_D_B), qd, ND);
        gemm(xd, P(l, V_D_W), P(l, V_D_B), vd, ND);

        fill_f32<<<nblk((long long)NGf, 256), 256, 0, stream>>>(outg, 0.f, (long long)NGf);
        fill_f32<<<nblk((long long)NDf, 256), 256, 0, stream>>>(outd, 0.f, (long long)NDf);

        relation(kg, vg, NG, qg, outg, NG, gg_src, gg_dst, Egg,
                 P(l, AREL_GG), P(l, MREL_GG), P(l, PREL_GG));
        relation(kd, vd, ND, qg, outg, NG, dg_src, dg_dst, Edg,
                 P(l, AREL_DG), P(l, MREL_DG), P(l, PREL_DG));
        relation(kg, vg, NG, qd, outd, ND, gd_src, gd_dst, Egd,
                 P(l, AREL_GD), P(l, MREL_GD), P(l, PREL_GD));

        gelu_ip_kernel<<<nblk((long long)NGf, 256), 256, 0, stream>>>(outg, (long long)NGf);
        gelu_ip_kernel<<<nblk((long long)NDf, 256), 256, 0, stream>>>(outd, (long long)NDf);
        gemm(outg, P(l, A_G_W), P(l, A_G_B), krel, NG);  // reuse krel as o_g
        gemm(outd, P(l, A_D_W), P(l, A_D_B), kd,   ND);  // reuse kd as o_d
        skip_combine_kernel<<<nblk((long long)NGf, 256), 256, 0, stream>>>(
            xg, krel, P(l, SKIP_G), (long long)NGf);
        skip_combine_kernel<<<nblk((long long)NDf, 256), 256, 0, stream>>>(
            xd, kd, P(l, SKIP_D), (long long)NDf);
    }

    // ---- output projection + pair gather ----
    gemm(xg, LIN_OUT_W, LIN_OUT_B, kg, NG);   // reuse kg as node_rep
    gather2_kernel<<<nblk((long long)NP * 32, 256), 256, 0, stream>>>(
        kg, nodea, nodeb, (float*)d_out, NP);

    (void)n_in; (void)out_size; (void)ws_size;
}
